// Upsampling_22668837388816
// MI455X (gfx1250) — matrix operand order repaired and checked by reading, compiled, not run on hardware
//
#include <hip/hip_runtime.h>
#include <hip/hip_bf16.h>

typedef __attribute__((ext_vector_type(16))) _Float16 v16h;
typedef __attribute__((ext_vector_type(8)))  float    v8f;
typedef __attribute__((ext_vector_type(4))) unsigned int uint4v;
typedef __attribute__((ext_vector_type(8))) int          int8v;
typedef __attribute__((ext_vector_type(4))) int          int4v;
typedef unsigned int u32;
typedef unsigned long long u64;

#define C_IN  64
#define C_OUT 64
#define H     128
#define W     128
#define OH    256
#define OW    256

// ---------------------------------------------------------------------------
// Kernel 1: fold 3x3 weights into per-parity 2x2 effective weights (f16),
// stored directly in WMMA A-fragment order:
//   idx = ((((parity*4 + ocb)*8 + kk)*32 + lane)*16 + j)
// A k order used by this code (16-bit A 16x32, contiguous):
//   lane<16: M=lane,    K = j
//   lane>=16: M=lane-16, K = j + 16
// K packs (ic, t, s) as K = ic*4 + t*2 + s  (t=row tap, s=col tap)
// ---------------------------------------------------------------------------
__global__ void build_weff(const float* __restrict__ w, _Float16* __restrict__ weff) {
    int idx  = blockIdx.x * 256 + threadIdx.x;      // 65536 total
    int j    = idx & 15;
    int lane = (idx >> 4) & 31;
    int kk   = (idx >> 9) & 7;
    int ocb  = (idx >> 12) & 3;
    int par  = (idx >> 14) & 3;
    int ry = par >> 1, rx = par & 1;

    int oc = ocb * 16 + (lane & 15);
    int K  = kk * 32 + j + ((lane >= 16) ? 16 : 0);
    int ic = K >> 2;
    int t  = (K >> 1) & 1;
    int s  = K & 1;

    // tap maps: ry=0 -> t0={ky0}, t1={ky1,ky2}; ry=1 -> t0={ky0,ky1}, t1={ky2}
    const float* wb = w + (size_t)(oc * C_IN + ic) * 9;
    float acc = 0.f;
#pragma unroll
    for (int ky = 0; ky < 3; ++ky) {
        int th = (ry == 0) ? (ky == 0 ? 0 : 1) : (ky == 2 ? 1 : 0);
        if (th != t) continue;
#pragma unroll
        for (int kx = 0; kx < 3; ++kx) {
            int sw = (rx == 0) ? (kx == 0 ? 0 : 1) : (kx == 2 ? 1 : 0);
            if (sw == s) acc += wb[ky * 3 + kx];
        }
    }
    weff[idx] = (_Float16)acc;
}

// ---------------------------------------------------------------------------
// Kernel 2: fused upsample(2x nearest) + 3x3 conv as 4 parity-class 2x2 convs,
// implicit GEMM with v_wmma_f32_16x16x32_f16.
//   block = 512 threads = 16 waves; wave -> (parity = wid>>2, ocb = wid&3)
//   block tile: one batch b, one input row iy, 16 input cols [ix0, ix0+16)
//   halo staging: TDM tensor_load_to_lds for interior blocks (one DMA op,
//   TENSORcnt), scalar fallback at top/left/bottom borders.
// ---------------------------------------------------------------------------
__global__ __launch_bounds__(512)
void upconv_wmma(const float* __restrict__ x, const _Float16* __restrict__ weff,
                 const float* __restrict__ bias, float* __restrict__ out) {
    __shared__ float xs[C_IN][3][18];   // packed: TDM fill order (x,y,z)

    const int ix0 = blockIdx.x * 16;
    const int iy  = blockIdx.y;
    const int b   = blockIdx.z;
    const int tid = threadIdx.x;
    const int lane = tid & 31;
    const int wid  = tid >> 5;              // 0..15

    const float* xb = x + (size_t)b * C_IN * H * W;

    const bool boundary = (iy == 0) | (iy == H - 1) | (ix0 == 0);
    if (boundary) {
        // ---- scalar halo fill with explicit zero padding ----
        for (int i = tid; i < C_IN * 3 * 18; i += 512) {
            int ic = i / 54;
            int r  = (i % 54) / 18;
            int cc = i % 18;
            int gy = iy - 1 + r;
            int gx = ix0 - 1 + cc;
            float v = 0.f;
            if ((unsigned)gy < (unsigned)H && (unsigned)gx < (unsigned)W)
                v = xb[((size_t)ic * H + gy) * W + gx];
            xs[ic][r][cc] = v;
        }
    } else if (wid == 0) {
        // ---- Tensor Data Mover: 3D tile (18 cols, 3 rows, 64 ch) -> LDS ----
        // D# per CDNA5 ISA. tensor_dim = remaining
        // extent from tile start, so right/bottom OOB reads return zero.
        const u64 gaddr = (u64)(uintptr_t)(xb + ((size_t)(iy - 1) * W + (ix0 - 1)));
        const u32 td0 = (u32)(129 - ix0);   // remaining cols from ix0-1
        const u32 td1 = (u32)(129 - iy);    // remaining rows from iy-1

        uint4v g0;
        g0[0] = 1u;                                   // count=1, user descriptor
        g0[1] = 0u;                                   // lds_addr: xs at LDS offset 0
        g0[2] = (u32)(gaddr & 0xffffffffull);         // global_addr[31:0]
        g0[3] = (u32)((gaddr >> 32) & 0x1ffffffu) | (2u << 30);  // [56:32] | type=2

        int8v g1;
        g1[0] = (int)(2u << 16);                      // data_size = 4 bytes
        g1[1] = (int)(td0 << 16);                     // tensor_dim0[15:0]
        g1[2] = (int)((td0 >> 16) | (td1 << 16));     // dim0 hi | tensor_dim1 lo
        g1[3] = (int)((td1 >> 16) | (18u << 16));     // dim1 hi | tile_dim0 = 18
        g1[4] = (int)(3u | (64u << 16));              // tile_dim1=3 | tile_dim2=64
        g1[5] = (int)W;                               // tensor_dim0_stride = 128
        g1[6] = (int)((u32)(H * W) << 16);            // tensor_dim1_stride lo16
        g1[7] = (int)((u32)(H * W) >> 16);            // tensor_dim1_stride hi
        int4v g2; g2[0] = C_IN; g2[1] = 0; g2[2] = 0; g2[3] = 0;  // tensor_dim2=64
        int4v g3; g3[0] = 0; g3[1] = 0; g3[2] = 0; g3[3] = 0;
        int8v g4 = {0, 0, 0, 0, 0, 0, 0, 0};          // extra group (clang-23
                                                      // 6-arg form): zero-filled

        __builtin_amdgcn_tensor_load_to_lds(g0, g1, g2, g3, g4, 0);
        __builtin_amdgcn_s_wait_tensorcnt(0);
    }
    __syncthreads();

    const int par = wid >> 2;               // parity class (ry,rx)
    const int ocb = wid & 3;                // oc block of 16
    const int ry = par >> 1, rx = par & 1;
    const int n  = lane & 15;               // N column (pixel in tile)
    const int hi = (lane >= 16) ? 1 : 0;

    // ---- load the 8 A fragments (fragment-ordered Weff, L2-resident) ----
    const v16h* ap = (const v16h*)weff + (size_t)((par * 4 + ocb) * 8) * 32 + lane;
    v16h a[8];
#pragma unroll
    for (int kk = 0; kk < 8; ++kk) a[kk] = ap[kk * 32];

    // ---- K loop: build B fragment from LDS (f32->f16), accumulate ----
    // B k order used here (matches A): lanes 0-15 hold K 0-15, lanes 16-31 hold
    // K 16-31; half j -> K = j + 16*hi, N = lane%16.
    v8f acc = {};
#pragma unroll
    for (int kk = 0; kk < 8; ++kk) {
        const int icb = kk * 8 + hi * 4;    // K = kk*32 + j + 16*hi ; ic = K>>2
        v16h bf;
#pragma unroll
        for (int j = 0; j < 16; ++j) {
            int ic = icb + (j >> 2);
            int t  = (j >> 1) & 1;
            int s  = j & 1;
            bf[j] = (_Float16)xs[ic][ry + t][n + rx + s];
        }
        acc = __builtin_amdgcn_wmma_f32_16x16x32_f16(
            /*neg_a=*/false, a[kk], /*neg_b=*/false, bf,
            /*c_mod=*/(short)0, acc, /*reuse_a=*/false, /*reuse_b=*/false);
    }

    // ---- add bias, scatter D (C/D layout: VGPR r, M = r + 8*hi, N = lane%16)
    const int oy = 2 * iy + ry;
    const int ox = 2 * (ix0 + n) + rx;
    float* ob = out + (size_t)b * C_OUT * OH * OW;
#pragma unroll
    for (int r = 0; r < 8; ++r) {
        int oc = ocb * 16 + r + hi * 8;
        ob[((size_t)oc * OH + oy) * OW + ox] = acc[r] + bias[oc];
    }
}

// ---------------------------------------------------------------------------
extern "C" void kernel_launch(void* const* d_in, const int* in_sizes, int n_in,
                              void* d_out, int out_size, void* d_ws, size_t ws_size,
                              hipStream_t stream) {
    const float* x    = (const float*)d_in[0];   // [B,64,128,128] f32
    const float* w    = (const float*)d_in[1];   // [64,64,3,3]    f32
    const float* bias = (const float*)d_in[2];   // [64]           f32
    float*       out  = (float*)d_out;           // [B,64,256,256] f32
    _Float16*    weff = (_Float16*)d_ws;         // 65536 halves = 128 KB

    int B = in_sizes[0] / (C_IN * H * W);

    // 1) fold weights into per-parity 2x2 f16 fragments
    build_weff<<<256, 256, 0, stream>>>(w, weff);

    // 2) fused upsample+conv via WMMA, TDM-staged halo tiles
    dim3 grid(W / 16, H, B);
    upconv_wmma<<<grid, 512, 0, stream>>>(x, weff, bias, out);
}
